// SelfPromptingHead_36335423324460
// MI455X (gfx1250) — matrix operand order repaired and checked by reading, not compiled, not run on hardware
//
#include <hip/hip_runtime.h>
#include <hip/hip_bf16.h>
#include <math.h>

// MI455X / gfx1250, wave32. Conv1+Conv2 as implicit GEMM with
// v_wmma_f32_16x16x32_bf16 (f32 accumulate), double-buffered LDS tiles
// filled with global_load_async_to_lds_b128 (ASYNCcnt) so tile DMA for
// chunk k+1 overlaps the WMMA chain for chunk k. Conv1 runs 2 M-tiles per
// wave (8 accumulators) to amortize B-fragment LDS reads (12 ds_load per 8 WMMAs).

typedef __attribute__((ext_vector_type(16))) __bf16 v16bf;
typedef __attribute__((ext_vector_type(8)))  float  v8f;

#define B_IMG 32
#define HW    64
#define SPATIAL (B_IMG * HW * HW)      // 131072 spatial positions
#define OUT_HW 256

// ---- CDNA5 async copy: per-lane 16B global -> LDS, tracked by ASYNCcnt ----
__device__ __forceinline__ void async_copy_b128(unsigned lds_off, const void* gptr) {
    asm volatile("global_load_async_to_lds_b128 %0, %1, off"
                 :: "v"(lds_off), "v"(gptr) : "memory");
}
__device__ __forceinline__ void wait_asynccnt0() {
    asm volatile("s_wait_asynccnt 0x0" ::: "memory");
}
__device__ __forceinline__ unsigned lds_addr(const void* p) {
    return (unsigned)(size_t)p;        // low 32 bits of generic ptr = LDS offset
}

// ---------------------------------------------------------------------------
// Pack x: NCHW fp32 -> NHWC bf16 (67MB, L2-resident afterwards)
// ---------------------------------------------------------------------------
__global__ __launch_bounds__(256)
void pack_x_nhwc(const float* __restrict__ x, __bf16* __restrict__ out) {
    int i = blockIdx.x * 256 + threadIdx.x;   // 32*64*64*256, exact multiple
    int c = i & 255;
    int xq = (i >> 8) & 63;
    int yq = (i >> 14) & 63;
    int b  = i >> 20;
    float v = x[(((size_t)b * 256 + c) * 64 + yq) * 64 + xq];
    out[i] = (__bf16)v;
}

// Pack weights: OIHW fp32 -> [tap][cout][cin] bf16 (K contiguous per row so
// B tiles can be async-DMA'd straight into LDS)
__global__ __launch_bounds__(256)
void pack_w(const float* __restrict__ wsrc, __bf16* __restrict__ dst, int Cin) {
    int i = blockIdx.x * 256 + threadIdx.x;   // 9*64*Cin, exact multiple
    int c   = i % Cin;
    int t2  = i / Cin;
    int o   = t2 & 63;
    int tap = t2 >> 6;
    dst[i] = (__bf16)wsrc[((size_t)o * Cin + c) * 9 + tap];
}

// ---------------------------------------------------------------------------
// Implicit-GEMM 3x3 conv + BN + ReLU with bf16 WMMA.
// Workgroup: 256 threads (8 waves), tile = (128*MPW) rows x 64 out-ch.
// Each wave: MPW x 16 rows x 64 ch -> MPW*4 v8f accumulators.
// FUSE=1: also applies the 1x1 conv3 (64->1) through LDS -> logits64.
// ---------------------------------------------------------------------------
template<int CIN, int FUSE, int MPW>
__global__ __launch_bounds__(256)
void conv3x3_bn_relu_wmma(const __bf16* __restrict__ in,     // NHWC bf16 [S][CIN]
                          const __bf16* __restrict__ wp,     // [9][64][CIN] bf16
                          const float* __restrict__ gg, const float* __restrict__ bb,
                          const float* __restrict__ mm, const float* __restrict__ vv,
                          __bf16* __restrict__ out,          // NHWC bf16 [S][64] (FUSE=0)
                          const float* __restrict__ w3,      // [64] fp32 (FUSE=1)
                          const float* __restrict__ b3,      // [1]  fp32 (FUSE=1)
                          float* __restrict__ lgout)         // [S]  fp32 (FUSE=1)
{
    static_assert(CIN % 32 == 0 && ((CIN / 32) & (CIN / 32 - 1)) == 0, "");
    constexpr int KCH  = CIN / 32;                // 32-wide K chunks per tap (pow2)
    constexpr int KSH  = (KCH == 8) ? 3 : 1;      // log2(KCH)
    constexpr int NCH  = 9 * KCH;                 // total K chunks
    constexpr int ROWS = 128 * MPW;               // spatial rows per workgroup

    __shared__ __bf16 lA[2][ROWS][40];            // double-buffered A tiles (+pad)
    __shared__ __bf16 lB[2][64][40];              // double-buffered B tiles [o][k]
    __shared__ float  lAct[FUSE ? 128 : 1][65];

    const int tid  = threadIdx.x;
    const int lane = tid & 31;
    const int wv   = tid >> 5;                    // 8 waves (wave32)
    const int sBase = blockIdx.x * ROWS;

    v8f acc[MPW][4];
    #pragma unroll
    for (int mh = 0; mh < MPW; ++mh)
        #pragma unroll
        for (int nt = 0; nt < 4; ++nt)
            #pragma unroll
            for (int e = 0; e < 8; ++e) acc[mh][nt][e] = 0.0f;

    // staging roles: each thread stages MPW half-rows of A + one 16B piece of B
    const int part = tid & 1;                     // 32B half of a 64B A row
    int rowSt[MPW], yrowA[MPW], xcolA[MPW], bimgA[MPW];
    #pragma unroll
    for (int p = 0; p < MPW; ++p) {
        rowSt[p] = (tid >> 1) + p * 128;
        const int sRow = sBase + rowSt[p];
        bimgA[p] = sRow >> 12;
        yrowA[p] = (sRow >> 6) & 63;
        xcolA[p] = sRow & 63;
    }
    const int bo = tid >> 2;                      // 0..63 (B out-channel row)
    const int bq = tid & 3;                       // 16B quarter of the 32-elem row

    auto stage = [&](int chunk, int buf) {
        const int tap = chunk >> KSH;
        const int c0  = (chunk & (KCH - 1)) * 32;
        const int ky  = (tap < 3) ? -1 : (tap < 6 ? 0 : 1);
        const int kx  = tap - (ky + 1) * 3 - 1;
        // ---- A tile: async DMA in-bounds rows, zero-fill border rows ----
        #pragma unroll
        for (int p = 0; p < MPW; ++p) {
            const int yy = yrowA[p] + ky, xx = xcolA[p] + kx;
            if ((unsigned)yy < (unsigned)HW && (unsigned)xx < (unsigned)HW) {
                const __bf16* src = in + (size_t)((bimgA[p] * HW + yy) * HW + xx) * CIN
                                       + c0 + part * 16;
                async_copy_b128(lds_addr(&lA[buf][rowSt[p]][part * 16]),     src);
                async_copy_b128(lds_addr(&lA[buf][rowSt[p]][part * 16 + 8]), src + 8);
            } else {
                uint4 z{};
                *(uint4*)&lA[buf][rowSt[p]][part * 16]     = z;
                *(uint4*)&lA[buf][rowSt[p]][part * 16 + 8] = z;
            }
        }
        // ---- B tile: [o][k] rows are contiguous in packed weights ----
        const __bf16* wsrc = wp + ((size_t)(tap * 64 + bo)) * CIN + c0 + bq * 8;
        async_copy_b128(lds_addr(&lB[buf][bo][bq * 8]), wsrc);
    };

    stage(0, 0);
    wait_asynccnt0();
    __syncthreads();

    for (int ch = 0; ch < NCH; ++ch) {
        const int buf = ch & 1;
        if (ch + 1 < NCH) stage(ch + 1, buf ^ 1);   // prefetch overlaps WMMA

        union U { v16bf v; uint4 q[2]; };
        // ---- A fragments (16x32 bf16, wave32 layout) ----
        // lanes 0-15: K 0..7 & 16..23 ; lanes 16-31: K 8..15 & 24..31
        const int khalf = (lane >> 4) << 3;
        U ua[MPW];
        #pragma unroll
        for (int mh = 0; mh < MPW; ++mh) {
            const int arow = wv * (16 * MPW) + mh * 16 + (lane & 15);
            ua[mh].q[0] = *(const uint4*)&lA[buf][arow][khalf];
            ua[mh].q[1] = *(const uint4*)&lA[buf][arow][16 + khalf];
        }
        // ---- all 4 B fragments, then back-to-back WMMAs ----
        const int kb = (lane >> 4) << 3;            // lanes 0-15: K 0..7 & 16..23, 16-31: K 8..15 & 24..31
        U ub[4];
        #pragma unroll
        for (int nt = 0; nt < 4; ++nt) {
            const int bcol = nt * 16 + (lane & 15);
            ub[nt].q[0] = *(const uint4*)&lB[buf][bcol][kb];
            ub[nt].q[1] = *(const uint4*)&lB[buf][bcol][16 + kb];
        }
        #pragma unroll
        for (int mh = 0; mh < MPW; ++mh)
            #pragma unroll
            for (int nt = 0; nt < 4; ++nt)
                acc[mh][nt] = __builtin_amdgcn_wmma_f32_16x16x32_bf16(
                    false, ua[mh].v, false, ub[nt].v, (short)0, acc[mh][nt], false, false);

        wait_asynccnt0();                           // next tile landed
        __syncthreads();                            // all waves done reading buf
    }

    // ---- epilogue: BN + ReLU (C/D layout: N = lane%16, M = vgpr + 8*(lane/16)) ----
    #pragma unroll
    for (int nt = 0; nt < 4; ++nt) {
        const int chn = nt * 16 + (lane & 15);
        const float sc = gg[chn] * rsqrtf(vv[chn] + 1e-5f);
        const float sh = bb[chn] - mm[chn] * sc;
        #pragma unroll
        for (int mh = 0; mh < MPW; ++mh) {
            #pragma unroll
            for (int r = 0; r < 8; ++r) {
                const int row = wv * (16 * MPW) + mh * 16 + ((lane >> 4) << 3) + r;
                const float val = fmaxf(acc[mh][nt][r] * sc + sh, 0.0f);
                if constexpr (FUSE) {
                    lAct[row][chn] = val;
                } else {
                    out[(size_t)(sBase + row) * 64 + chn] = (__bf16)val;
                }
            }
        }
    }

    if constexpr (FUSE) {                           // fused 1x1 conv3 -> logits
        __syncthreads();
        if (tid < 128) {
            float sum = b3[0];
            #pragma unroll
            for (int c = 0; c < 64; ++c) sum += lAct[tid][c] * w3[c];
            lgout[sBase + tid] = sum;
        }
    }
}

// ---------------------------------------------------------------------------
// Bilinear resize 64x64 -> 256x256 (half-pixel centers) + sigmoid
// ---------------------------------------------------------------------------
__global__ __launch_bounds__(256)
void resize_sigmoid_kernel(const float* __restrict__ lg,
                           float* __restrict__ heat, float* __restrict__ lgout)
{
    int i = blockIdx.x * 256 + threadIdx.x;   // 32*256*256, exact multiple
    int b  = i >> 16;
    int p  = i & 65535;
    int oy = p >> 8, ox = p & 255;
    float sy = (oy + 0.5f) * 0.25f - 0.5f;
    float sx = (ox + 0.5f) * 0.25f - 0.5f;
    int y0 = (int)floorf(sy), x0 = (int)floorf(sx);
    float fy = sy - (float)y0, fx = sx - (float)x0;
    int y0c = min(max(y0, 0), 63),     y1c = min(max(y0 + 1, 0), 63);
    int x0c = min(max(x0, 0), 63),     x1c = min(max(x0 + 1, 0), 63);
    const float* src = lg + (size_t)b * 4096;
    float v00 = src[y0c * 64 + x0c], v01 = src[y0c * 64 + x1c];
    float v10 = src[y1c * 64 + x0c], v11 = src[y1c * 64 + x1c];
    float v0 = v00 + (v01 - v00) * fx;
    float v1 = v10 + (v11 - v10) * fx;
    float val = v0 + (v1 - v0) * fy;
    lgout[i] = val;
    heat[i]  = 1.0f / (1.0f + expf(-val));
}

// ---------------------------------------------------------------------------
// Point sampling: 9x9-NMS peaks, top-3 positives (argmax fallback),
// 2 hash-random negatives among low-response pixels; valid-first compaction.
// One workgroup per batch image.
// ---------------------------------------------------------------------------
__global__ __launch_bounds__(256)
void sample_points_kernel(const float* __restrict__ heat,
                          float* __restrict__ coords, int* __restrict__ labels)
{
    const int b = blockIdx.x;
    const float* hm = heat + (size_t)b * (OUT_HW * OUT_HW);

    __shared__ float sPv[256 * 3]; __shared__ int sPi[256 * 3];
    __shared__ float sAv[256];     __shared__ int sAi[256];
    __shared__ float sNv[256 * 2]; __shared__ int sNi[256 * 2];

    float tv[3] = {-1e30f, -1e30f, -1e30f}; int ti[3] = {-1, -1, -1};
    float nv[2] = {-1e30f, -1e30f};         int ni[2] = {-1, -1};
    float av = -1e30f; int ai = 0;

    for (int p = threadIdx.x; p < OUT_HW * OUT_HW; p += 256) {
        const float v = hm[p];
        if (v > av) { av = v; ai = p; }
        if (v > 0.1f) {                       // PEAK_THR; 9x9 local-max (MIN_DIST=8 -> 9)
            const int y = p >> 8, x = p & 255;
            bool pk = true;
            for (int dy = -4; dy <= 4; ++dy) {
                const int yy = y + dy;
                if ((unsigned)yy > 255u) continue;
                for (int dx = -4; dx <= 4; ++dx) {
                    const int xx = x + dx;
                    if ((unsigned)xx > 255u) continue;
                    if (hm[(yy << 8) + xx] > v) { pk = false; }
                }
                if (!pk) break;
            }
            if (pk) {
                for (int j = 0; j < 3; ++j) {
                    if (v > tv[j] || (v == tv[j] && p < ti[j])) {
                        for (int k = 2; k > j; --k) { tv[k] = tv[k-1]; ti[k] = ti[k-1]; }
                        tv[j] = v; ti[j] = p; break;
                    }
                }
            }
        }
        if (v < 0.3f) {                       // LOW_THR negatives: hashed uniform score
            unsigned h = (unsigned)(p + 1) * 2654435761u ^ ((unsigned)(b + 1) * 0x9E3779B9u);
            h ^= h >> 16; h *= 0x85EBCA6Bu; h ^= h >> 13;
            const float r = (float)(h >> 8) * (1.0f / 16777216.0f);
            for (int j = 0; j < 2; ++j) {
                if (r > nv[j]) {
                    for (int k = 1; k > j; --k) { nv[k] = nv[k-1]; ni[k] = ni[k-1]; }
                    nv[j] = r; ni[j] = p; break;
                }
            }
        }
    }
    for (int j = 0; j < 3; ++j) { sPv[threadIdx.x * 3 + j] = tv[j]; sPi[threadIdx.x * 3 + j] = ti[j]; }
    for (int j = 0; j < 2; ++j) { sNv[threadIdx.x * 2 + j] = nv[j]; sNi[threadIdx.x * 2 + j] = ni[j]; }
    sAv[threadIdx.x] = av; sAi[threadIdx.x] = ai;
    __syncthreads();

    if (threadIdx.x == 0) {
        // global argmax
        float gav = -1e30f; int gai = 0;
        for (int i = 0; i < 256; ++i)
            if (sAv[i] > gav) { gav = sAv[i]; gai = sAi[i]; }
        // global top-3 peaks
        float pv[3]; int pi[3]; bool pvld[3];
        for (int s = 0; s < 3; ++s) {
            float best = -1e30f; int bi = -1;
            for (int i = 0; i < 256 * 3; ++i) {
                const int id = sPi[i];
                if (id < 0) continue;
                bool used = false;
                for (int t = 0; t < s; ++t) if (pi[t] == id) used = true;
                if (used) continue;
                const float vv = sPv[i];
                if (vv > best || (vv == best && (bi < 0 || id < bi))) { best = vv; bi = id; }
            }
            pv[s] = best; pi[s] = bi;
            pvld[s] = (bi >= 0) && (best > 0.1f);
        }
        if (!pvld[0]) { pi[0] = gai; }        // argmax fallback; slot 0 always valid
        pvld[0] = true;
        // global top-2 negatives, excluding valid positives
        int qi[2]; bool qvld[2];
        for (int s = 0; s < 2; ++s) {
            float best = -1e30f; int bi = -1;
            for (int i = 0; i < 256 * 2; ++i) {
                const int id = sNi[i];
                if (id < 0) continue;
                bool bad = false;
                for (int t = 0; t < 3; ++t) if (pvld[t] && pi[t] == id) bad = true;
                for (int t = 0; t < s; ++t) if (qi[t] == id) bad = true;
                if (bad) continue;
                if (sNv[i] > best) { best = sNv[i]; bi = id; }
            }
            qi[s] = bi; qvld[s] = (bi >= 0);
        }
        // compose candidates and compact valid-first (stable)
        int cx[5], cy[5], cl[5]; bool cv[5];
        for (int s = 0; s < 3; ++s) {
            cx[s] = pvld[s] ? (pi[s] & 255) : 0;
            cy[s] = pvld[s] ? (pi[s] >> 8)  : 0;
            cv[s] = pvld[s]; cl[s] = 1;
        }
        for (int s = 0; s < 2; ++s) {
            cx[3 + s] = qvld[s] ? (qi[s] & 255) : 0;
            cy[3 + s] = qvld[s] ? (qi[s] >> 8)  : 0;
            cv[3 + s] = qvld[s]; cl[3 + s] = 0;
        }
        float* cc = coords + (size_t)b * 10;
        int*   ll = labels + (size_t)b * 5;
        int w = 0;
        for (int k = 0; k < 5; ++k)
            if (cv[k]) { cc[w * 2] = (float)cx[k]; cc[w * 2 + 1] = (float)cy[k]; ll[w] = cl[k]; ++w; }
        for (int k = 0; k < 5; ++k)
            if (!cv[k]) { cc[w * 2] = 0.0f; cc[w * 2 + 1] = 0.0f; ll[w] = -1; ++w; }
    }
}

// ---------------------------------------------------------------------------
extern "C" void kernel_launch(void* const* d_in, const int* in_sizes, int n_in,
                              void* d_out, int out_size, void* d_ws, size_t ws_size,
                              hipStream_t stream) {
    const float* x  = (const float*)d_in[0];
    const float* W1 = (const float*)d_in[1];
    const float* g1 = (const float*)d_in[2];
    const float* b1 = (const float*)d_in[3];
    const float* m1 = (const float*)d_in[4];
    const float* v1 = (const float*)d_in[5];
    const float* W2 = (const float*)d_in[6];
    const float* g2 = (const float*)d_in[7];
    const float* b2 = (const float*)d_in[8];
    const float* m2 = (const float*)d_in[9];
    const float* v2 = (const float*)d_in[10];
    const float* W3 = (const float*)d_in[11];
    const float* b3 = (const float*)d_in[12];
    // d_in[13], d_in[14] = output_h/w (fixed at 256)

    // workspace layout (all offsets 256B aligned), ~85 MB total
    char* w = (char*)d_ws;
    __bf16* xpack = (__bf16*)w;  w += (size_t)SPATIAL * 256 * 2;   // 67,108,864
    __bf16* wp1   = (__bf16*)w;  w += (size_t)9 * 64 * 256 * 2;    //    294,912
    __bf16* wp2   = (__bf16*)w;  w += (size_t)9 * 64 * 64 * 2;     //     73,728
    __bf16* act1  = (__bf16*)w;  w += (size_t)SPATIAL * 64 * 2;    // 16,777,216
    float*  lg64  = (float*)w;                                     //    524,288

    // output layout: heatmap | coords | labels | logits
    float* heat   = (float*)d_out;                   // 32*1*256*256
    float* coords = heat + 32 * OUT_HW * OUT_HW;     // 32*1*5*2
    int*   labels = (int*)(coords + 32 * 5 * 2);     // 32*1*5
    float* lgout  = (float*)(labels + 32 * 5);       // 32*1*256*256

    pack_x_nhwc<<<(SPATIAL * 256) / 256, 256, 0, stream>>>(x, xpack);
    pack_w<<<(9 * 64 * 256) / 256, 256, 0, stream>>>(W1, wp1, 256);
    pack_w<<<(9 * 64 * 64) / 256, 256, 0, stream>>>(W2, wp2, 64);

    // conv1: 2 M-tiles/wave (256 rows per WG), conv2: 1 M-tile/wave + fused head
    conv3x3_bn_relu_wmma<256, 0, 2><<<SPATIAL / 256, 256, 0, stream>>>(
        xpack, wp1, g1, b1, m1, v1, act1, nullptr, nullptr, nullptr);
    conv3x3_bn_relu_wmma<64, 1, 1><<<SPATIAL / 128, 256, 0, stream>>>(
        act1, wp2, g2, b2, m2, v2, nullptr, W3, b3, lg64);

    resize_sigmoid_kernel<<<(32 * OUT_HW * OUT_HW) / 256, 256, 0, stream>>>(lg64, heat, lgout);
    sample_points_kernel<<<32, 256, 0, stream>>>(heat, coords, labels);
}
